// PinnLayer_13030930776227
// MI455X (gfx1250) — matrix operand order repaired and checked by reading, not compiled, not run on hardware
//
#include <hip/hip_runtime.h>

typedef __attribute__((ext_vector_type(16))) _Float16 v16h;
typedef __attribute__((ext_vector_type(8)))  float    v8f;

#define NPTS 65536
#define NB   32
#define HDIM 256
#define TMAXF 20.0f
#define RB_ROWS 16
#define NUM_RB (NPTS / RB_ROWS)      /* 4096 row blocks of 16 samples */
#define W1FRAG_HALVES (HDIM * HDIM)  /* 65536 f16 = 128 KB */
#define W2FRAG_HALVES (HDIM * NB)    /* 8192 f16  = 16 KB  */
#define WFRAG_HALVES  (W1FRAG_HALVES + W2FRAG_HALVES)

// ---------------------------------------------------------------------------
// Contiguous-k B-fragment layout used here with V_WMMA_F32_16X16X32_F16 (B is 32x16, K x N):
//   lane = n | (k & 16);  element idx within v16h = k & 15
// A-fragment layout (A is 16x32, M x K):
//   lane = m + (k & 16);  idx = k & 15
// C/D (f32 16x16): lane holds column n = lane&15; vgpr v holds row
//   m = v + 8*(lane>>4)
// ---------------------------------------------------------------------------

__global__ __launch_bounds__(256) void prep_weights(
    const float* __restrict__ W1, const float* __restrict__ W2,
    _Float16* __restrict__ ws) {
  int e = blockIdx.x * blockDim.x + threadIdx.x;
  if (e < HDIM * HDIM) {
    int k = e >> 8, n = e & 255;                 // W1[k][n], z1 = h0 @ W1
    int kt = k >> 5, kl = k & 31, nt = n >> 4, nl = n & 15;
    int lane = nl | (kl & 16);
    int idx  = kl & 15;
    ws[(((kt * 16) + nt) * 32 + lane) * 16 + idx] = (_Float16)W1[e];
  } else if (e < HDIM * HDIM + HDIM * NB) {
    int e2 = e - HDIM * HDIM;
    int k = e2 >> 5, n = e2 & 31;                // W2[k][n]
    int kt = k >> 5, kl = k & 31, nt = n >> 4, nl = n & 15;
    int lane = nl | (kl & 16);
    int idx  = kl & 15;
    ws[W1FRAG_HALVES + (((kt * 2) + nt) * 32 + lane) * 16 + idx] =
        (_Float16)W2[e2];
  }
}

__global__ __launch_bounds__(256) void pinn_fused(
    const float* __restrict__ t_in, const float* __restrict__ p_in,
    const float* __restrict__ W0,  const float* __restrict__ b0,
    const float* __restrict__ b1,  const float* __restrict__ b2,
    const float* __restrict__ lm,  const float* __restrict__ ld,
    const float* __restrict__ lb,  const float* __restrict__ bwi,
    const _Float16* __restrict__ wf, float* __restrict__ out) {
  // Whole swizzled W1+W2 fragment image cached in LDS (144 KB).
  __shared__ __align__(32) _Float16 wS[WFRAG_HALVES];
  // A-fragment staging for {h, h', h''} of layer-0 and layer-1 outputs.
  __shared__ __align__(32) _Float16 h0A[3][8][32][16];   // 24 KB
  __shared__ __align__(32) _Float16 h1A[3][8][32][16];   // 24 KB
  __shared__ float outS[3][16][32];                      // 6 KB: out/out'/out''
  __shared__ float W0s[HDIM], b0s[HDIM], b1s[HDIM];      // 3 KB
  __shared__ float b2s[NB], lmS[NB], ldS[NB], bwS[NB];   // 0.5 KB
  __shared__ float lbS[NB][NB];                          // 4 KB

  const int tid  = threadIdx.x;
  const int lane = tid & 31;
  const int wave = tid >> 5;

  // ---- one-time parameter + weight-fragment load ----
  W0s[tid] = W0[tid];
  b0s[tid] = b0[tid];
  b1s[tid] = b1[tid];
  if (tid < NB) {
    b2s[tid] = b2[tid];
    lmS[tid] = lm[tid];
    ldS[tid] = ld[tid];
    bwS[tid] = bwi[tid];
  }
  for (int i = tid; i < NB * NB; i += 256) lbS[i >> 5][i & 31] = lb[i];
  {
    const v16h* __restrict__ src = (const v16h*)wf;
    v16h* dst = (v16h*)&wS[0];
#pragma unroll 2
    for (int i = tid; i < WFRAG_HALVES / 16; i += 256) dst[i] = src[i];
  }
  __syncthreads();

  for (int rb = blockIdx.x; rb < NUM_RB; rb += gridDim.x) {
    const int row0 = rb * RB_ROWS;

    // ---- step 1: wave w builds K-tile w of the h0 / h0' / h0'' A-fragments
    {
      const int kt = wave;
      const int m  = lane & 15;
      const float t = t_in[row0 + m];
      const float x = 2.0f * t / TMAXF - 1.0f;   // normalised input
      const float c = 2.0f / TMAXF;              // dx/dt
      v16h a0, a1, a2;
#pragma unroll
      for (int idx = 0; idx < 16; ++idx) {
        const int kl = idx | ((lane >> 4) << 4);
        const int k  = kt * 32 + kl;
        const float w   = W0s[k];
        const float h   = tanhf(x * w + b0s[k]);
        const float a   = c * w;
        const float hp  = (1.0f - h * h) * a;
        const float hpp = -2.0f * h * hp * a;
        a0[idx] = (_Float16)h;
        a1[idx] = (_Float16)hp;
        a2[idx] = (_Float16)hpp;
      }
      *(v16h*)&h0A[0][kt][lane][0] = a0;
      *(v16h*)&h0A[1][kt][lane][0] = a1;
      *(v16h*)&h0A[2][kt][lane][0] = a2;
    }
    __syncthreads();

    // ---- step 2: layer-1 GEMM (3 matrices share B); each wave: 2 N-tiles
#pragma unroll
    for (int rep = 0; rep < 2; ++rep) {
      const int nt = wave + rep * 8;
      v8f acc0 = {}, acc1 = {}, acc2 = {};
#pragma unroll
      for (int kt = 0; kt < 8; ++kt) {
        const v16h A0 = *(const v16h*)&h0A[0][kt][lane][0];
        const v16h A1 = *(const v16h*)&h0A[1][kt][lane][0];
        const v16h A2 = *(const v16h*)&h0A[2][kt][lane][0];
        const v16h B  = *(const v16h*)&wS[(((kt * 16) + nt) * 32 + lane) * 16];
        acc0 = __builtin_amdgcn_wmma_f32_16x16x32_f16(false, A0, false, B,
                                                      (short)0, acc0, false, false);
        acc1 = __builtin_amdgcn_wmma_f32_16x16x32_f16(false, A1, false, B,
                                                      (short)0, acc1, false, false);
        acc2 = __builtin_amdgcn_wmma_f32_16x16x32_f16(false, A2, false, B,
                                                      (short)0, acc2, false, false);
      }
      // tanh + chain rule, relayout D-tile -> layer-2 A-fragments
      const int col  = nt * 16 + (lane & 15);    // hidden index (layer-2 K)
      const float bias = b1s[col];
      const int kt2 = col >> 5;
      const int kl2 = col & 31;
      const int lane2b = kl2 & 16;
      const int idx2   = kl2 & 15;
#pragma unroll
      for (int v = 0; v < 8; ++v) {
        const int m = v + ((lane >> 4) << 3);    // sample row within block
        const float h1  = tanhf(acc0[v] + bias);
        const float zp  = acc1[v];
        const float zpp = acc2[v];
        const float s    = 1.0f - h1 * h1;
        const float h1p  = s * zp;
        const float h1pp = s * zpp - 2.0f * h1 * h1p * zp;
        const int lane2 = m + lane2b;
        h1A[0][kt2][lane2][idx2] = (_Float16)h1;
        h1A[1][kt2][lane2][idx2] = (_Float16)h1p;
        h1A[2][kt2][lane2][idx2] = (_Float16)h1pp;
      }
    }
    __syncthreads();

    // ---- step 3: layer-2; 6 independent (matrix, N-tile) GEMMs on waves 0..5
    if (wave < 6) {
      const int mat = wave >> 1;   // 0: out, 1: out', 2: out''
      const int nt  = wave & 1;
      v8f acc = {};
#pragma unroll
      for (int kt = 0; kt < 8; ++kt) {
        const v16h A = *(const v16h*)&h1A[mat][kt][lane][0];
        const v16h B = *(const v16h*)&wS[W1FRAG_HALVES +
                                         (((kt * 2) + nt) * 32 + lane) * 16];
        acc = __builtin_amdgcn_wmma_f32_16x16x32_f16(false, A, false, B,
                                                     (short)0, acc, false, false);
      }
      const int col = nt * 16 + (lane & 15);     // bus index
      const float bias = (mat == 0) ? b2s[col] : 0.0f;
#pragma unroll
      for (int v = 0; v < 8; ++v) {
        const int m = v + ((lane >> 4) << 3);
        const float val = acc[v] + bias;
        outS[mat][m][col] = val;
        if (mat < 2)
          out[(size_t)mat * NPTS * NB + (size_t)(row0 + m) * NB + col] = val;
      }
    }
    __syncthreads();

    // ---- step 4: physics residual with sin coupling
#pragma unroll
    for (int rep = 0; rep < 2; ++rep) {
      const int e = tid + rep * 256;             // 512 = 16 rows x 32 buses
      const int m = e >> 5;
      const int i = e & 31;
      const float di = outS[0][m][i];
      float con = 0.0f;
#pragma unroll
      for (int j = 0; j < NB; ++j)
        con += lbS[i][j] * __sinf(di - outS[0][m][j]);
      const float phys = lmS[i] * bwS[i] * outS[2][m][i] +
                         ldS[i] * outS[1][m][i] + con -
                         p_in[(size_t)(row0 + m) * NB + i];
      out[(size_t)2 * NPTS * NB + (size_t)(row0 + m) * NB + i] = phys;
    }
    // next iteration's first shared write (h0A) is fenced by the barrier
    // after step 2, and outS/h1A rewrites are fenced by the barriers above,
    // so no trailing barrier is required.
  }
}

extern "C" void kernel_launch(void* const* d_in, const int* in_sizes, int n_in,
                              void* d_out, int out_size, void* d_ws,
                              size_t ws_size, hipStream_t stream) {
  const float* t_in = (const float*)d_in[0];   // input_time  (N,1)
  const float* p_in = (const float*)d_in[1];   // input_power (N,NB)
  const float* W0   = (const float*)d_in[2];
  const float* b0   = (const float*)d_in[3];
  const float* W1   = (const float*)d_in[4];
  const float* b1   = (const float*)d_in[5];
  const float* W2   = (const float*)d_in[6];
  const float* b2   = (const float*)d_in[7];
  const float* lm   = (const float*)d_in[8];
  const float* ld   = (const float*)d_in[9];
  const float* lb   = (const float*)d_in[10];
  const float* bwi  = (const float*)d_in[11];
  _Float16* wf = (_Float16*)d_ws;              // 144 KB fragment cache
  float* out = (float*)d_out;                  // [out | out_t | physics]

  prep_weights<<<(HDIM * HDIM + HDIM * NB + 255) / 256, 256, 0, stream>>>(
      W1, W2, wf);
  pinn_fused<<<512, 256, 0, stream>>>(t_in, p_in, W0, b0, b1, b2, lm, ld, lb,
                                      bwi, wf, out);
}
